// DyRep_62904091018094
// MI455X (gfx1250) — matrix operand order repaired and checked by reading, not compiled, not run on hardware
//
#include <hip/hip_runtime.h>
#include <hip/hip_bf16.h>
#include <math.h>

// Problem constants (match reference)
constexpr int Nn    = 8192;
constexpr int Hd    = 128;
constexpr int Bb    = 64;
constexpr int Ssamp = 20;
constexpr int MAXNB = 256;

typedef __attribute__((ext_vector_type(16))) _Float16 v16h;
typedef __attribute__((ext_vector_type(8)))  float    v8f;

// Hardware-native transcendentals (v_exp_f32 / v_rcp_f32 TRANS ops)
__device__ __forceinline__ float fast_exp(float x) {
  return __builtin_amdgcn_exp2f(x * 1.44269504088896340736f);
}
__device__ __forceinline__ float fast_sigmoid(float x) {
  return __builtin_amdgcn_rcpf(1.0f + fast_exp(-x));
}

// ---------------------------------------------------------------------------
// rate(z_a, z_b, k) = psi_k * log1p(exp(clip(g/psi_k, -75, 75)))
// g = 0.5 * (za.W[:H] + zb.W[H:] + zb.W[:H] + za.W[H:]) + b_k
// ---------------------------------------------------------------------------
__device__ __forceinline__ float rate_fn(const float* __restrict__ za,
                                         const float* __restrict__ zb,
                                         int kk,
                                         const float* __restrict__ psi,
                                         const float* __restrict__ W_om,
                                         const float* __restrict__ b_om) {
  const float* W = W_om + kk * 2 * Hd;
  float s1 = 0.f, s2 = 0.f;
  for (int h = 0; h < Hd; ++h) {
    s1 += za[h] * W[h] + zb[h] * W[Hd + h];
    s2 += zb[h] * W[h] + za[h] * W[Hd + h];
  }
  float p = psi[kk];
  float g = 0.5f * (s1 + s2) + b_om[kk];
  float x = g / p;
  x = fminf(75.f, fmaxf(-75.f, x));
  return p * log1pf(expf(x));
}

// ---------------------------------------------------------------------------
// init: copy embeddings -> z (output region), last_event_time -> workspace.
// Re-run every launch so graph replay is deterministic.
// ---------------------------------------------------------------------------
__global__ void init_kernel(const float* __restrict__ emb,
                            const float* __restrict__ let_in,
                            float* __restrict__ z,
                            float* __restrict__ let_ws) {
  size_t i = blockIdx.x * (size_t)blockDim.x + threadIdx.x;
  size_t stride = gridDim.x * (size_t)blockDim.x;
  for (size_t j = i; j < (size_t)Nn * Hd; j += stride) z[j] = emb[j];
  for (size_t j = i; j < (size_t)Nn; j += stride) let_ws[j] = let_in[j];
}

// ---------------------------------------------------------------------------
// lam[b] = rate(emb[u[b]], emb[v[b]], k[b])
// ---------------------------------------------------------------------------
__global__ void lam_kernel(const int* __restrict__ u, const int* __restrict__ v,
                           const int* __restrict__ k,
                           const float* __restrict__ emb,
                           const float* __restrict__ psi,
                           const float* __restrict__ W_om,
                           const float* __restrict__ b_om,
                           float* __restrict__ out) {
  int b = threadIdx.x;
  if (b < Bb) {
    out[b] = rate_fn(emb + (size_t)u[b] * Hd, emb + (size_t)v[b] * Hd,
                     k[b], psi, W_om, b_om);
  }
}

// ---------------------------------------------------------------------------
// L_surv[s] = sum_b (2*(ru0+ru1) + rv1) / SS   (deterministic tree reduce)
// ---------------------------------------------------------------------------
__global__ void surv_kernel(const int* __restrict__ u, const int* __restrict__ v,
                            const int* __restrict__ uo, const int* __restrict__ vo,
                            const float* __restrict__ emb,
                            const float* __restrict__ psi,
                            const float* __restrict__ W_om,
                            const float* __restrict__ b_om,
                            float* __restrict__ out) {
  __shared__ float red[Bb];
  int s = blockIdx.x;
  int b = threadIdx.x;
  const float* zu  = emb + (size_t)u[b] * Hd;
  const float* zv  = emb + (size_t)v[b] * Hd;
  const float* zvo = emb + (size_t)vo[b * Ssamp + s] * Hd;
  const float* zuo = emb + (size_t)uo[b * Ssamp + s] * Hd;
  float ru0 = rate_fn(zu, zvo, 0, psi, W_om, b_om);
  float ru1 = rate_fn(zu, zvo, 1, psi, W_om, b_om);
  float rv1 = rate_fn(zv, zuo, 1, psi, W_om, b_om);
  red[b] = 2.f * (ru0 + ru1) + rv1;
  __syncthreads();
  for (int off = 32; off > 0; off >>= 1) {
    if (b < off) red[b] += red[b + off];
    __syncthreads();
  }
  if (b == 0) out[s] = red[0] / (float)Ssamp;
}

// ---------------------------------------------------------------------------
// Sequential per-event scan. Single workgroup, 256 threads = 8 waves.
// Exploits A-sparsity: computes h = W_h @ z[j] + b_h only for neighbor rows,
// batched 16 rows at a time through v_wmma_f32_16x16x32_f16.
// The B operand (W_h^T tiles) is constant -> gathered ONCE into registers:
// hot loop per K-step is 2x ds_load_b128 (A frag) + v_wmma.
// Sigmoid/max tail is branchless (vector q loads + selects, no exec branches).
// ---------------------------------------------------------------------------
__global__ void __launch_bounds__(256)
scan_kernel(const int* __restrict__ ue, const int* __restrict__ ve,
            const float* __restrict__ te,
            const float* __restrict__ A, const float* __restrict__ S,
            const float* __restrict__ W_S, const float* __restrict__ W_R,
            const float* __restrict__ W_t, const float* __restrict__ W_h,
            const float* __restrict__ b_h,
            float* __restrict__ z, float* __restrict__ let) {
  __shared__ __align__(16) _Float16 sZ[16][Hd];  // neighbor z tile (A operand)
  __shared__ __align__(16) float s_part[256];
  __shared__ __align__(16) float s_hs[2][Hd];
  __shared__ __align__(16) float s_zrow[2][Hd];
  __shared__ __align__(16) float s_new[2][Hd];
  __shared__ __align__(16) float s_q[16];
  __shared__ int   s_nbr[MAXNB];
  __shared__ int   s_cnt;
  __shared__ float s_qsum;
  __shared__ int   s_u, s_v;
  __shared__ float s_t;
  __shared__ float s_dt[2];

  const int tid  = threadIdx.x;
  const int lane = tid & 31;     // wave32
  const int wv   = tid >> 5;     // 8 waves -> 8 N-tiles of 16 cols

  // Persistent per-lane operands: output column + bias + 4 B fragments.
  const int ncol  = wv * 16 + (lane & 15);
  const int koff  = (lane >> 4) * 16;
  const int mbase = 8 * (lane >> 4);   // C rows covered by this lane
  const float bh  = b_h[ncol];
  v16h Bfrag[4];
  #pragma unroll
  for (int kk = 0; kk < 4; ++kk) {
    const float* wrow = W_h + (size_t)ncol * Hd + kk * 32 + koff;
    #pragma unroll
    for (int j = 0; j < 16; ++j)
      Bfrag[kk][j] = (_Float16)wrow[j];   // contiguous b128 global loads
  }

  for (int e = 0; e < Bb; ++e) {
    if (tid == 0) { s_u = ue[e]; s_v = ve[e]; s_t = te[e]; }
    __syncthreads();
    const int uu = s_u, vv = s_v;
    const float tt = s_t;
    if (tid == 0) {
      s_dt[0] = tt - let[uu];   // side 0: neighbors of v, dt from let[u]
      s_dt[1] = tt - let[vv];   // side 1: neighbors of u, dt from let[v]
    }

    for (int side = 0; side < 2; ++side) {
      const int n2 = (side == 0) ? vv : uu;
      const float* Arow = A + (size_t)n2 * Nn;
      const float* Srow = S + (size_t)n2 * Nn;
      if (tid == 0) s_cnt = 0;
      for (int i = tid; i < Hd; i += 256) s_hs[side][i] = -1.0e30f;
      __syncthreads();

      // Neighbor discovery (one 32KB A-row scan) + qsum partials
      float part = 0.f;
      for (int j = tid; j < Nn; j += 256) {
        if (Arow[j] > 0.f) {
          int idx = atomicAdd(&s_cnt, 1);
          if (idx < MAXNB) s_nbr[idx] = j;
          part += fast_exp(Srow[j]);
        }
      }
      s_part[tid] = part;
      __syncthreads();
      for (int off = 128; off > 0; off >>= 1) {   // deterministic tree reduce
        if (tid < off) s_part[tid] += s_part[tid + off];
        __syncthreads();
      }
      if (tid == 0) s_qsum = s_part[0];
      __syncthreads();
      int cnt = s_cnt; if (cnt > MAXNB) cnt = MAXNB;
      const float qinv = __builtin_amdgcn_rcpf(s_qsum + 1e-7f);

      // Process neighbors in chunks of 16 rows via WMMA
      for (int c0 = 0; c0 < cnt; c0 += 16) {
        int nrows = cnt - c0; if (nrows > 16) nrows = 16;
        for (int idx = tid; idx < 16 * Hd; idx += 256) {
          int m = idx >> 7, h = idx & 127;
          float val = 0.f;
          if (m < nrows) val = z[(size_t)s_nbr[c0 + m] * Hd + h];
          sZ[m][h] = (_Float16)val;
        }
        if (tid < 16)
          s_q[tid] = (tid < nrows) ? (fast_exp(Srow[s_nbr[c0 + tid]]) * qinv) : 0.f;
        __syncthreads();

        // C[16 x 16] per wave: rows = neighbors, cols = wv*16..wv*16+15
        v8f c = {};
        #pragma unroll
        for (int kk = 0; kk < 4; ++kk) {
          v16h a;
          // A frag (16x32 f16): one contiguous 16-half group per lane
          {
            int m = lane & 15;
            int base = kk * 32 + ((lane >> 4) * 16);
            #pragma unroll
            for (int j = 0; j < 16; ++j) {
              a[j]     = sZ[m][base + j];
            }
          }
          c = __builtin_amdgcn_wmma_f32_16x16x32_f16(
              false, a, false, Bfrag[kk], (short)0, c, false, false);
        }

        // Branchless sigmoid(q_m * (h + b_h)) + max over this chunk's rows.
        // q loaded unconditionally as 2x ds_load_b128; validity via select.
        {
          float4 q0 = *(const float4*)&s_q[mbase];
          float4 q1 = *(const float4*)&s_q[mbase + 4];
          float qr[8] = {q0.x, q0.y, q0.z, q0.w, q1.x, q1.y, q1.z, q1.w};
          float mx = -1.0e30f;
          #pragma unroll
          for (int r = 0; r < 8; ++r) {
            float val = fast_sigmoid(qr[r] * (c[r] + bh));
            mx = fmaxf(mx, (mbase + r < nrows) ? val : -1.0e30f);
          }
          s_part[tid] = mx;
        }
        __syncthreads();
        if (tid < Hd) {   // combine the two lane-halves covering each column
          int w2 = tid >> 4, l2 = tid & 15;
          float m2 = fmaxf(s_part[w2 * 32 + l2], s_part[w2 * 32 + l2 + 16]);
          s_hs[side][tid] = fmaxf(s_hs[side][tid], m2);
        }
        __syncthreads();
      }
      if (tid < Hd && cnt == 0) s_hs[side][tid] = 0.f;   // mask.any() == False
      __syncthreads();
    }

    // Snapshot pre-update rows z[v], z[u] (Sterm uses the carry state)
    if (tid < Hd)            s_zrow[0][tid]      = z[(size_t)vv * Hd + tid];
    else if (tid < 2 * Hd)   s_zrow[1][tid - Hd] = z[(size_t)uu * Hd + (tid - Hd)];
    __syncthreads();

    // new[side][i] = sigmoid(hs_side.W_S[i,:] + zrow_side.W_R[i,:] + dt*W_t[i])
    // float4 form: b128 global loads for weight rows, ds_load_b128 for LDS.
    {
      int side = tid >> 7;
      int i = tid & 127;
      const float4* wsrow = (const float4*)(W_S + (size_t)i * Hd);
      const float4* wrrow = (const float4*)(W_R + (size_t)i * Hd);
      const float4* hsv   = (const float4*)s_hs[side];
      const float4* zrv   = (const float4*)s_zrow[side];
      float acc = 0.f;
      #pragma unroll 4
      for (int h4 = 0; h4 < Hd / 4; ++h4) {
        float4 ws = wsrow[h4], wr = wrrow[h4];
        float4 hs = hsv[h4],   zr = zrv[h4];
        acc += hs.x * ws.x + zr.x * wr.x;
        acc += hs.y * ws.y + zr.y * wr.y;
        acc += hs.z * ws.z + zr.z * wr.z;
        acc += hs.w * ws.w + zr.w * wr.w;
      }
      acc += s_dt[side] * W_t[i];
      s_new[side][i] = fast_sigmoid(acc);
    }
    __syncthreads();
    // z.at[[v,u]].set(...): write v first, then u (u wins if aliased)
    if (tid < Hd) z[(size_t)vv * Hd + tid] = s_new[0][tid];
    __syncthreads();
    if (tid < Hd) z[(size_t)uu * Hd + tid] = s_new[1][tid];
    if (tid == 0) { let[uu] = tt; let[vv] = tt; }
    __syncthreads();

    // gfx1250 prefetch of next event's A rows (global_prefetch_b8)
    if (e + 1 < Bb && tid == 0) {
      __builtin_prefetch(A + (size_t)ue[e + 1] * Nn, 0, 1);
      __builtin_prefetch(A + (size_t)ve[e + 1] * Nn, 0, 1);
    }
  }
}

// ---------------------------------------------------------------------------
extern "C" void kernel_launch(void* const* d_in, const int* in_sizes, int n_in,
                              void* d_out, int out_size, void* d_ws, size_t ws_size,
                              hipStream_t stream) {
  const int*   u      = (const int*)d_in[0];
  const int*   v      = (const int*)d_in[1];
  const float* t      = (const float*)d_in[2];
  const int*   k      = (const int*)d_in[3];
  const int*   uo     = (const int*)d_in[4];
  const int*   vo     = (const int*)d_in[5];
  const float* A      = (const float*)d_in[6];
  const float* S      = (const float*)d_in[7];
  const float* emb    = (const float*)d_in[8];
  const float* let_in = (const float*)d_in[9];
  const float* W_S    = (const float*)d_in[10];
  const float* W_R    = (const float*)d_in[11];
  const float* W_t    = (const float*)d_in[12];
  const float* W_h    = (const float*)d_in[13];
  const float* b_h    = (const float*)d_in[14];
  const float* psi    = (const float*)d_in[15];
  const float* W_om   = (const float*)d_in[16];
  const float* b_om   = (const float*)d_in[17];

  float* out    = (float*)d_out;
  float* lam    = out;                // [64]
  float* Lsurv  = out + Bb;           // [20]
  float* zf     = out + Bb + Ssamp;   // [N*H]
  float* let_ws = (float*)d_ws;       // [N]

  init_kernel<<<512, 256, 0, stream>>>(emb, let_in, zf, let_ws);
  lam_kernel<<<1, Bb, 0, stream>>>(u, v, k, emb, psi, W_om, b_om, lam);
  surv_kernel<<<Ssamp, Bb, 0, stream>>>(u, v, uo, vo, emb, psi, W_om, b_om, Lsurv);
  scan_kernel<<<1, 256, 0, stream>>>(u, v, t, A, S, W_S, W_R, W_t, W_h, b_h,
                                     zf, let_ws);
}
